// NeuralCompressor_54812372632232
// MI455X (gfx1250) — compile-verified, matrix operand order checked by reading, not run on hardware
//
#include <hip/hip_runtime.h>

// ---------------------------------------------------------------------------
// MI455X / gfx1250: bf16 WMMA GEMMs (f32 accumulate) for encoder/decoder MLPs
// and the residual-VQ score GEMM with fused argmin. wave32, 256-thr blocks.
// GEMM uses double-buffered LDS tiles with async global->LDS (ASYNCcnt);
// register-staged overlap as fallback. Weights pre-transposed to [N][K] so
// both A and B tiles stage with contiguous 16B chunks (no LDS scatter).
// ---------------------------------------------------------------------------

typedef __attribute__((ext_vector_type(16))) __bf16 bf16x16;
typedef __attribute__((ext_vector_type(8)))  float  f32x8;
typedef int v4i_vec __attribute__((__vector_size__(16)));

#define TOK    8192        // B*S = 4*2048
#define DIN    1024
#define DHID   2048
#define DBOT   256
#define NCODE  8192
#define NQLEV  4
#define LN_EPS 1e-5f

#if defined(__has_builtin)
#if __has_builtin(__builtin_amdgcn_global_load_async_to_lds_b128) && \
    __has_builtin(__builtin_amdgcn_s_wait_asynccnt)
#define HAVE_ASYNC_LDS 1
#endif
#endif
#ifndef HAVE_ASYNC_LDS
#define HAVE_ASYNC_LDS 0
#endif

#define AS_GLOBAL __attribute__((address_space(1)))
#define AS_LDS    __attribute__((address_space(3)))

#if HAVE_ASYNC_LDS
__device__ __forceinline__ void async_copy_b128(const __bf16* g, __bf16* l)
{
  __builtin_amdgcn_global_load_async_to_lds_b128(
      (AS_GLOBAL v4i_vec*)g, (AS_LDS v4i_vec*)l, 0, 0);
}
#endif

// --------------------------- WMMA fragment load ----------------------------
// 16-bit A-matrix 16x32 layout (ISA 7.12.2): lane L holds row m = L&15;
// lanes 0-15 hold K = {0..7, 16..23}, lanes 16-31 hold K = {8..15, 24..31}.
// Same swizzle used for B stored as [n][k].
__device__ __forceinline__ bf16x16 load_frag(const __bf16* __restrict__ base, int ldm)
{
  int lane = threadIdx.x & 31;
  int m    = lane & 15;
  int kh   = (lane >> 4) << 3;
  const __bf16* p = base + (size_t)m * ldm;
  bf16x16 f;
#pragma unroll
  for (int i = 0; i < 4; ++i) {
    f[2*i]     = p[kh + 2*i];
    f[2*i + 1] = p[kh + 2*i + 1];
  }
#pragma unroll
  for (int i = 0; i < 4; ++i) {
    f[8 + 2*i]     = p[16 + kh + 2*i];
    f[8 + 2*i + 1] = p[16 + kh + 2*i + 1];
  }
  return f;
}

// ------------------------------- GEMM ---------------------------------------
// C[M,N] = A[M,K] @ BT[N,K]^T. A/BT bf16 row-major, C f32. 128x128x32 tiles,
// double-buffered LDS. 8 waves (2x4); each wave computes 64x32 (4x2 tiles).
__global__ void __launch_bounds__(256) gemm_bf16_wmma(
    const __bf16* __restrict__ A, const __bf16* __restrict__ BT,
    float* __restrict__ C, int M, int N, int K)
{
  __shared__ __bf16 sA[2][128 * 32];
  __shared__ __bf16 sB[2][128 * 32];

  const int tid  = threadIdx.x;
  const int warp = tid >> 5;
  const int lane = tid & 31;
  const int wr   = warp >> 2;       // 0..1
  const int wc   = warp & 3;        // 0..3
  const int bm   = blockIdx.y * 128;
  const int bn   = blockIdx.x * 128;
  const int r0   = tid >> 2;        // 0..63
  const int g8   = (tid & 3) * 8;   // k-chunk within 32

  const __bf16* gA0 = A  + (size_t)(bm + r0)      * K + g8;
  const __bf16* gA1 = A  + (size_t)(bm + r0 + 64) * K + g8;
  const __bf16* gB0 = BT + (size_t)(bn + r0)      * K + g8;
  const __bf16* gB1 = BT + (size_t)(bn + r0 + 64) * K + g8;
  __bf16* lA0;  __bf16* lA1;  __bf16* lB0;  __bf16* lB1;

  f32x8 acc[4][2];
#pragma unroll
  for (int rt = 0; rt < 4; ++rt)
#pragma unroll
    for (int ct = 0; ct < 2; ++ct)
#pragma unroll
      for (int r = 0; r < 8; ++r) acc[rt][ct][r] = 0.0f;

  auto ldsPtrs = [&](int buf) {
    lA0 = &sA[buf][r0 * 32 + g8];
    lA1 = &sA[buf][(r0 + 64) * 32 + g8];
    lB0 = &sB[buf][r0 * 32 + g8];
    lB1 = &sB[buf][(r0 + 64) * 32 + g8];
  };
  auto computeTile = [&](int buf) {
    bf16x16 afr[4], bfr[2];
#pragma unroll
    for (int rt = 0; rt < 4; ++rt)
      afr[rt] = load_frag(&sA[buf][(wr * 64 + rt * 16) * 32], 32);
#pragma unroll
    for (int ct = 0; ct < 2; ++ct)
      bfr[ct] = load_frag(&sB[buf][(wc * 32 + ct * 16) * 32], 32);
#pragma unroll
    for (int rt = 0; rt < 4; ++rt)
#pragma unroll
      for (int ct = 0; ct < 2; ++ct)
        acc[rt][ct] = __builtin_amdgcn_wmma_f32_16x16x32_bf16(
            false, afr[rt], false, bfr[ct], (short)0, acc[rt][ct], false, false);
  };

  const int T = K >> 5;  // K/32 tiles

#if HAVE_ASYNC_LDS
  // ---- async global->LDS double buffering (ASYNCcnt) ----
  ldsPtrs(0);
  async_copy_b128(gA0, lA0); async_copy_b128(gA1, lA1);
  async_copy_b128(gB0, lB0); async_copy_b128(gB1, lB1);
  for (int t = 0; t < T; ++t) {
    if (t + 1 < T) {
      ldsPtrs((t + 1) & 1);
      int kk = (t + 1) << 5;
      async_copy_b128(gA0 + kk, lA0); async_copy_b128(gA1 + kk, lA1);
      async_copy_b128(gB0 + kk, lB0); async_copy_b128(gB1 + kk, lB1);
      __builtin_amdgcn_s_wait_asynccnt(4);   // tile t landed; t+1 in flight
    } else {
      __builtin_amdgcn_s_wait_asynccnt(0);
    }
    __syncthreads();
    computeTile(t & 1);
    __syncthreads();
  }
#else
  // ---- register-staged double buffering ----
  uint4 ra0, ra1, rb0, rb1;
  ra0 = *(const uint4*)gA0; ra1 = *(const uint4*)gA1;
  rb0 = *(const uint4*)gB0; rb1 = *(const uint4*)gB1;
  ldsPtrs(0);
  *(uint4*)lA0 = ra0; *(uint4*)lA1 = ra1;
  *(uint4*)lB0 = rb0; *(uint4*)lB1 = rb1;
  for (int t = 0; t < T; ++t) {
    if (t + 1 < T) {
      int kk = (t + 1) << 5;
      ra0 = *(const uint4*)(gA0 + kk); ra1 = *(const uint4*)(gA1 + kk);
      rb0 = *(const uint4*)(gB0 + kk); rb1 = *(const uint4*)(gB1 + kk);
      if (t + 2 < T) {  // speculative prefetch two tiles ahead
        __builtin_prefetch(gA0 + ((t + 2) << 5), 0, 1);
        __builtin_prefetch(gB0 + ((t + 2) << 5), 0, 1);
      }
    }
    __syncthreads();
    computeTile(t & 1);            // overlaps with in-flight global loads
    if (t + 1 < T) {
      ldsPtrs((t + 1) & 1);
      *(uint4*)lA0 = ra0; *(uint4*)lA1 = ra1;
      *(uint4*)lB0 = rb0; *(uint4*)lB1 = rb1;
    }
  }
#endif

  // C/D layout: VGPR r, lanes 0-15 -> (m=r, n=lane); lanes 16-31 -> (m=8+r, n=lane-16)
  const int lh = lane >> 4, ln = lane & 15;
  const int m0 = bm + wr * 64, n0 = bn + wc * 32;
#pragma unroll
  for (int rt = 0; rt < 4; ++rt)
#pragma unroll
    for (int ct = 0; ct < 2; ++ct) {
      size_t rowb = (size_t)(m0 + rt * 16 + 8 * lh);
      int col = n0 + ct * 16 + ln;
#pragma unroll
      for (int r = 0; r < 8; ++r)
        C[(rowb + r) * N + col] = acc[rt][ct][r];
    }
}

// ---------------------- LayerNorm (+bias in, optional GELU) -----------------
__global__ void __launch_bounds__(256) ln_gelu_kernel(
    const float* __restrict__ in, const float* __restrict__ bias,
    const float* __restrict__ gma, const float* __restrict__ bta,
    float* __restrict__ outA, float* __restrict__ outB,
    __bf16* __restrict__ outH, int H, int doGelu)
{
  __shared__ float red[256];
  const int row = blockIdx.x;
  const int tid = threadIdx.x;
  const int items = H >> 8;  // H in {256,1024,2048} -> 1..8
  float xs[8];
  float sum = 0.f, sq = 0.f;
  for (int i = 0; i < items; ++i) {
    int idx = tid + (i << 8);
    float x = in[(size_t)row * H + idx] + bias[idx];
    xs[i] = x; sum += x; sq += x * x;
  }
  red[tid] = sum; __syncthreads();
  for (int s = 128; s > 0; s >>= 1) { if (tid < s) red[tid] += red[tid + s]; __syncthreads(); }
  float S = red[0]; __syncthreads();
  red[tid] = sq; __syncthreads();
  for (int s = 128; s > 0; s >>= 1) { if (tid < s) red[tid] += red[tid + s]; __syncthreads(); }
  float Q = red[0];
  float mean = S / (float)H;
  float var  = Q / (float)H - mean * mean;
  float inv  = rsqrtf(var + LN_EPS);
  for (int i = 0; i < items; ++i) {
    int idx = tid + (i << 8);
    float y = (xs[i] - mean) * inv * gma[idx] + bta[idx];
    if (doGelu) y = 0.5f * y * (1.0f + erff(y * 0.70710678118654752f));
    size_t o = (size_t)row * H + idx;
    if (outA) outA[o] = y;
    if (outB) outB[o] = y;
    if (outH) outH[o] = (__bf16)y;
  }
}

// ----------------------- fused VQ score GEMM + argmin -----------------------
// 16 tokens per block; each of 8 waves scans NCODE/8 = 1024 codes.
// dist(n) = ||c_n||^2 - 2 * <r, c_n>  (scores via WMMA bf16).
__global__ void __launch_bounds__(256) vq_argmin_kernel(
    const __bf16* __restrict__ resb,   // [TOK][DBOT]
    const __bf16* __restrict__ cb,     // [NCODE][DBOT]  (already [n][k])
    const float*  __restrict__ cnorm,  // [NCODE]
    int* __restrict__ idx_out)         // [TOK]
{
  __shared__ __bf16 sR[16 * DBOT];
  __shared__ float  sMin[8][16];
  __shared__ int    sIdx[8][16];

  const int tid  = threadIdx.x;
  const int wave = tid >> 5;
  const int lane = tid & 31;
  const int tokenBase = blockIdx.x * 16;

  {  // 16*256 bf16 = 512 16B chunks, 2 per thread
    const uint4* src = (const uint4*)(resb + (size_t)tokenBase * DBOT);
    uint4* dst = (uint4*)sR;
    dst[tid] = src[tid];
    dst[tid + 256] = src[tid + 256];
  }
  __syncthreads();

  bf16x16 af[8];
#pragma unroll
  for (int ks = 0; ks < 8; ++ks) af[ks] = load_frag(sR + ks * 32, DBOT);

  float minv[8]; int mini[8];
#pragma unroll
  for (int r = 0; r < 8; ++r) { minv[r] = 3.4e38f; mini[r] = 0; }

  const int colSeg = wave * (NCODE / 8);
  for (int t = 0; t < (NCODE / 8) / 16; ++t) {
    int colbase = colSeg + t * 16;
    if (t + 1 < (NCODE / 8) / 16)  // prefetch next codebook tile (L2-resident)
      __builtin_prefetch(cb + (size_t)(colbase + 16) * DBOT + (lane & 15) * DBOT, 0, 1);
    f32x8 acc;
#pragma unroll
    for (int r = 0; r < 8; ++r) acc[r] = 0.0f;
#pragma unroll
    for (int ks = 0; ks < 8; ++ks) {
      bf16x16 bf = load_frag(cb + (size_t)colbase * DBOT + ks * 32, DBOT);
      acc = __builtin_amdgcn_wmma_f32_16x16x32_bf16(
          false, af[ks], false, bf, (short)0, acc, false, false);
    }
    int n = colbase + (lane & 15);
    float cn = cnorm[n];
#pragma unroll
    for (int r = 0; r < 8; ++r) {
      float d = cn - 2.0f * acc[r];
      if (d < minv[r]) { minv[r] = d; mini[r] = n; }
    }
  }

  // reduce across the 16 lanes of each half-wave (rows differ between halves)
#pragma unroll
  for (int off = 8; off >= 1; off >>= 1) {
#pragma unroll
    for (int r = 0; r < 8; ++r) {
      float ov = __shfl_xor(minv[r], off, 32);
      int   oi = __shfl_xor(mini[r], off, 32);
      if (ov < minv[r] || (ov == minv[r] && oi < mini[r])) { minv[r] = ov; mini[r] = oi; }
    }
  }
  if ((lane & 15) == 0) {
    int half = lane >> 4;
#pragma unroll
    for (int r = 0; r < 8; ++r) {
      sMin[wave][half * 8 + r] = minv[r];
      sIdx[wave][half * 8 + r] = mini[r];
    }
  }
  __syncthreads();
  if (tid < 16) {
    float bv = sMin[0][tid]; int bi = sIdx[0][tid];
#pragma unroll
    for (int w = 1; w < 8; ++w) {
      float v = sMin[w][tid]; int i = sIdx[w][tid];
      if (v < bv || (v == bv && i < bi)) { bv = v; bi = i; }
    }
    idx_out[tokenBase + tid] = bi;
  }
}

// ----------------- VQ gather / residual update / loss accumulate ------------
__global__ void __launch_bounds__(256) vq_update_kernel(
    const int* __restrict__ idx, const float* __restrict__ cb_f32,
    float* __restrict__ residual, float* __restrict__ qtot,
    __bf16* __restrict__ resb, float* __restrict__ idx_out_f,
    float* __restrict__ loss_acc)
{
  const int wave = threadIdx.x >> 5;
  const int lane = threadIdx.x & 31;
  const int t = blockIdx.x * 8 + wave;
  const int ci = idx[t];
  const float* cp = cb_f32 + (size_t)ci * DBOT;
  float* rp = residual + (size_t)t * DBOT;
  float* qp = qtot + (size_t)t * DBOT;
  __bf16* rb = resb + (size_t)t * DBOT;
  float lsum = 0.f;
#pragma unroll
  for (int j = 0; j < 8; ++j) {
    int d = lane + 32 * j;
    float c = cp[d];
    float r = rp[d];
    float diff = r - c;
    lsum += diff * diff;
    qp[d] += c;
    rp[d] = diff;
    rb[d] = (__bf16)diff;
  }
#pragma unroll
  for (int off = 16; off >= 1; off >>= 1) lsum += __shfl_xor(lsum, off, 32);
  if (lane == 0) {
    atomicAdd(loss_acc, lsum);
    idx_out_f[t] = (float)ci;
  }
}

// ------------------------------ small utils ---------------------------------
__global__ void cast_f32_bf16_kernel(const float* __restrict__ in,
                                     __bf16* __restrict__ out, size_t n)
{
  size_t i = (size_t)blockIdx.x * blockDim.x + threadIdx.x;
  size_t stride = (size_t)gridDim.x * blockDim.x;
  for (; i < n; i += stride) out[i] = (__bf16)in[i];
}

// in[R][C] f32 -> out[C][R] bf16 (32x32 LDS tiles, coalesced both sides)
__global__ void __launch_bounds__(256) transpose_cast_kernel(
    const float* __restrict__ in, __bf16* __restrict__ out, int R, int Ccols)
{
  __shared__ __bf16 tile[32][33];
  const int bx = blockIdx.x * 32;   // col base
  const int by = blockIdx.y * 32;   // row base
  const int tx = threadIdx.x & 31;
  const int ty = threadIdx.x >> 5;  // 0..7
#pragma unroll
  for (int i = 0; i < 32; i += 8)
    tile[ty + i][tx] = (__bf16)in[(size_t)(by + ty + i) * Ccols + bx + tx];
  __syncthreads();
#pragma unroll
  for (int i = 0; i < 32; i += 8)
    out[(size_t)(bx + ty + i) * R + by + tx] = tile[tx][ty + i];
}

__global__ void zero_f32_kernel(float* __restrict__ p, size_t n)
{
  size_t i = (size_t)blockIdx.x * blockDim.x + threadIdx.x;
  size_t stride = (size_t)gridDim.x * blockDim.x;
  for (; i < n; i += stride) p[i] = 0.0f;
}

__global__ void cnorm_kernel(const float* __restrict__ cb, float* __restrict__ cn)
{
  const int wave = threadIdx.x >> 5;
  const int lane = threadIdx.x & 31;
  const int code = blockIdx.x * 8 + wave;
  const float* p = cb + (size_t)code * DBOT;
  float s = 0.f;
#pragma unroll
  for (int j = 0; j < 8; ++j) { float v = p[lane + 32 * j]; s += v * v; }
#pragma unroll
  for (int off = 16; off >= 1; off >>= 1) s += __shfl_xor(s, off, 32);
  if (lane == 0) cn[code] = s;
}

__global__ void finalize_quant_kernel(const float* __restrict__ qtot,
                                      float* __restrict__ qout,
                                      __bf16* __restrict__ qbf, size_t n)
{
  size_t i = (size_t)blockIdx.x * blockDim.x + threadIdx.x;
  size_t stride = (size_t)gridDim.x * blockDim.x;
  for (; i < n; i += stride) { float q = qtot[i]; qout[i] = q; qbf[i] = (__bf16)q; }
}

__global__ void bias_add_kernel(const float* __restrict__ in,
                                const float* __restrict__ bias,
                                float* __restrict__ out, int H, size_t n)
{
  size_t i = (size_t)blockIdx.x * blockDim.x + threadIdx.x;
  size_t stride = (size_t)gridDim.x * blockDim.x;
  for (; i < n; i += stride) out[i] = in[i] + bias[i % H];
}

__global__ void finalize_loss_kernel(const float* __restrict__ acc,
                                     float* __restrict__ out)
{
  if (threadIdx.x == 0 && blockIdx.x == 0)
    out[0] = 0.25f * acc[0] / (float)((size_t)TOK * DBOT);
}

// ------------------------------- launcher ------------------------------------
static inline size_t alignUp256(size_t x) { return (x + 255) & ~(size_t)255; }

extern "C" void kernel_launch(void* const* d_in, const int* in_sizes, int n_in,
                              void* d_out, int out_size, void* d_ws, size_t ws_size,
                              hipStream_t stream)
{
  (void)in_sizes; (void)n_in; (void)out_size; (void)ws_size;
  const float* x    = (const float*)d_in[0];
  const float* w1   = (const float*)d_in[1];
  const float* b1   = (const float*)d_in[2];
  const float* g1   = (const float*)d_in[3];
  const float* be1  = (const float*)d_in[4];
  const float* w2   = (const float*)d_in[5];
  const float* b2   = (const float*)d_in[6];
  const float* g2   = (const float*)d_in[7];
  const float* be2  = (const float*)d_in[8];
  const float* w3   = (const float*)d_in[9];
  const float* b3   = (const float*)d_in[10];
  const float* g3   = (const float*)d_in[11];
  const float* be3  = (const float*)d_in[12];
  const float* dw1  = (const float*)d_in[13];
  const float* db1  = (const float*)d_in[14];
  const float* dg1  = (const float*)d_in[15];
  const float* dbe1 = (const float*)d_in[16];
  const float* dw2  = (const float*)d_in[17];
  const float* db2  = (const float*)d_in[18];
  const float* dg2  = (const float*)d_in[19];
  const float* dbe2 = (const float*)d_in[20];
  const float* dw3  = (const float*)d_in[21];
  const float* db3  = (const float*)d_in[22];
  const float* cbs  = (const float*)d_in[23];   // [NQ][NCODE][DBOT]

  float* out_z     = (float*)d_out;                                   // TOK*DBOT
  float* out_q     = out_z + (size_t)TOK * DBOT;                      // TOK*DBOT
  float* out_idx   = out_q + (size_t)TOK * DBOT;                      // NQ*TOK
  float* out_rec   = out_idx + (size_t)NQLEV * TOK;                   // TOK*DIN
  float* out_loss  = out_rec + (size_t)TOK * DIN;                     // 1

  char* base = (char*)d_ws;
  size_t off = 0;
  auto alloc = [&](size_t bytes) { void* p = base + off; off = alignUp256(off + bytes); return p; };

  // weights pre-transposed to [N][K] bf16
  __bf16* wb1t  = (__bf16*)alloc((size_t)DHID * DIN * 2);   // [DHID][DIN]
  __bf16* wb2t  = (__bf16*)alloc((size_t)DIN * DHID * 2);   // [DIN][DHID]
  __bf16* wb3t  = (__bf16*)alloc((size_t)DBOT * DIN * 2);   // [DBOT][DIN]
  __bf16* dwb1t = (__bf16*)alloc((size_t)DIN * DBOT * 2);   // [DIN][DBOT]
  __bf16* dwb2t = (__bf16*)alloc((size_t)DHID * DIN * 2);   // [DHID][DIN]
  __bf16* dwb3t = (__bf16*)alloc((size_t)DIN * DHID * 2);   // [DIN][DHID]
  __bf16* xb    = (__bf16*)alloc((size_t)TOK * DIN * 2);
  __bf16* cbb   = (__bf16*)alloc((size_t)NQLEV * NCODE * DBOT * 2);
  float*  actBig  = (float*)alloc((size_t)TOK * DHID * 4);
  __bf16* actBigB = (__bf16*)alloc((size_t)TOK * DHID * 2);
  float*  actMid  = (float*)alloc((size_t)TOK * DIN * 4);
  __bf16* actMidB = (__bf16*)alloc((size_t)TOK * DIN * 2);
  float*  resid   = (float*)alloc((size_t)TOK * DBOT * 4);
  __bf16* residB  = (__bf16*)alloc((size_t)TOK * DBOT * 2);
  float*  qtot    = (float*)alloc((size_t)TOK * DBOT * 4);
  __bf16* qbf     = (__bf16*)alloc((size_t)TOK * DBOT * 2);
  float*  cnorm   = (float*)alloc((size_t)NQLEV * NCODE * 4);
  int*    idxbuf  = (int*)alloc((size_t)NQLEV * TOK * 4);
  float*  lossAcc = (float*)alloc(256);

  const dim3 blk(256);
  const dim3 castGrid(2048);

  // 0) zero accumulators (harness does not re-poison between replays)
  hipLaunchKernelGGL(zero_f32_kernel, dim3(512), blk, 0, stream, qtot, (size_t)TOK * DBOT);
  hipLaunchKernelGGL(zero_f32_kernel, dim3(1), blk, 0, stream, lossAcc, (size_t)1);

  // 1) casts / transposed casts to bf16
  hipLaunchKernelGGL(cast_f32_bf16_kernel, castGrid, blk, 0, stream, x,   xb,  (size_t)TOK * DIN);
  hipLaunchKernelGGL(cast_f32_bf16_kernel, castGrid, blk, 0, stream, cbs, cbb, (size_t)NQLEV * NCODE * DBOT);
  hipLaunchKernelGGL(transpose_cast_kernel, dim3(DHID/32, DIN/32),  blk, 0, stream, w1,  wb1t,  DIN,  DHID);
  hipLaunchKernelGGL(transpose_cast_kernel, dim3(DIN/32,  DHID/32), blk, 0, stream, w2,  wb2t,  DHID, DIN);
  hipLaunchKernelGGL(transpose_cast_kernel, dim3(DBOT/32, DIN/32),  blk, 0, stream, w3,  wb3t,  DIN,  DBOT);
  hipLaunchKernelGGL(transpose_cast_kernel, dim3(DIN/32,  DBOT/32), blk, 0, stream, dw1, dwb1t, DBOT, DIN);
  hipLaunchKernelGGL(transpose_cast_kernel, dim3(DHID/32, DIN/32),  blk, 0, stream, dw2, dwb2t, DIN,  DHID);
  hipLaunchKernelGGL(transpose_cast_kernel, dim3(DIN/32,  DHID/32), blk, 0, stream, dw3, dwb3t, DHID, DIN);

  // 2) codebook squared norms (all levels)
  hipLaunchKernelGGL(cnorm_kernel, dim3(NQLEV * NCODE / 8), blk, 0, stream, cbs, cnorm);

  // 3) encoder
  hipLaunchKernelGGL(gemm_bf16_wmma, dim3(DHID / 128, TOK / 128), blk, 0, stream,
                     xb, wb1t, actBig, TOK, DHID, DIN);
  hipLaunchKernelGGL(ln_gelu_kernel, dim3(TOK), blk, 0, stream,
                     actBig, b1, g1, be1, (float*)nullptr, (float*)nullptr, actBigB, DHID, 1);
  hipLaunchKernelGGL(gemm_bf16_wmma, dim3(DIN / 128, TOK / 128), blk, 0, stream,
                     actBigB, wb2t, actMid, TOK, DIN, DHID);
  hipLaunchKernelGGL(ln_gelu_kernel, dim3(TOK), blk, 0, stream,
                     actMid, b2, g2, be2, (float*)nullptr, (float*)nullptr, actMidB, DIN, 1);
  hipLaunchKernelGGL(gemm_bf16_wmma, dim3(DBOT / 128, TOK / 128), blk, 0, stream,
                     actMidB, wb3t, actMid /* z_pre, reuse */, TOK, DBOT, DIN);
  hipLaunchKernelGGL(ln_gelu_kernel, dim3(TOK), blk, 0, stream,
                     actMid, b3, g3, be3, out_z, resid, residB, DBOT, 0);

  // 4) residual VQ, 4 levels
  for (int q = 0; q < NQLEV; ++q) {
    const __bf16* cbq  = cbb + (size_t)q * NCODE * DBOT;
    const float*  cbqf = cbs + (size_t)q * NCODE * DBOT;
    const float*  cnq  = cnorm + (size_t)q * NCODE;
    int* idxq = idxbuf + (size_t)q * TOK;
    hipLaunchKernelGGL(vq_argmin_kernel, dim3(TOK / 16), blk, 0, stream,
                       residB, cbq, cnq, idxq);
    hipLaunchKernelGGL(vq_update_kernel, dim3(TOK / 8), blk, 0, stream,
                       idxq, cbqf, resid, qtot, residB, out_idx + (size_t)q * TOK, lossAcc);
  }

  // 5) quantized = quant_total (STE is value-identity); bf16 for decoder
  hipLaunchKernelGGL(finalize_quant_kernel, castGrid, blk, 0, stream,
                     qtot, out_q, qbf, (size_t)TOK * DBOT);

  // 6) decoder
  hipLaunchKernelGGL(gemm_bf16_wmma, dim3(DIN / 128, TOK / 128), blk, 0, stream,
                     qbf, dwb1t, actMid, TOK, DIN, DBOT);
  hipLaunchKernelGGL(ln_gelu_kernel, dim3(TOK), blk, 0, stream,
                     actMid, db1, dg1, dbe1, (float*)nullptr, (float*)nullptr, actMidB, DIN, 1);
  hipLaunchKernelGGL(gemm_bf16_wmma, dim3(DHID / 128, TOK / 128), blk, 0, stream,
                     actMidB, dwb2t, actBig, TOK, DHID, DIN);
  hipLaunchKernelGGL(ln_gelu_kernel, dim3(TOK), blk, 0, stream,
                     actBig, db2, dg2, dbe2, (float*)nullptr, (float*)nullptr, actBigB, DHID, 1);
  hipLaunchKernelGGL(gemm_bf16_wmma, dim3(DIN / 128, TOK / 128), blk, 0, stream,
                     actBigB, dwb3t, actMid, TOK, DIN, DHID);
  hipLaunchKernelGGL(bias_add_kernel, castGrid, blk, 0, stream,
                     actMid, db3, out_rec, DIN, (size_t)TOK * DIN);

  // 7) commitment loss
  hipLaunchKernelGGL(finalize_loss_kernel, dim3(1), dim3(1), 0, stream, lossAcc, out_loss);
}
